// SpectralMeshGraphNet_45827301048784
// MI455X (gfx1250) — matrix operand order repaired and checked by reading, compiled, not run on hardware
//
#include <hip/hip_runtime.h>
#include <hip/hip_bf16.h>

// ---------------------------------------------------------------------------
// SpectralMeshGraphNet on MI455X (gfx1250), wave32 + WMMA f32_16x16x32_f16.
// One wave (block=32) owns a 32-row tile (2 M-subtiles) so every B weight
// fragment loaded from memory feeds two WMMAs: gather/concat -> GEMM1 (WMMA)
// -> relu -> LDS (f16) -> GEMM2 (WMMA) -> LayerNorm -> residual ->
// store (+ optional f32 atomic segment-sum scatter).
// ---------------------------------------------------------------------------

typedef __attribute__((ext_vector_type(16))) _Float16 v16h;
typedef __attribute__((ext_vector_type(8)))  float    v8f;

#define WMMA(A,B,C) __builtin_amdgcn_wmma_f32_16x16x32_f16(false,(A),false,(B),(short)0,(C),false,false)

// A-fragment K offset (contiguous, matches B): lanes 0-15 hold
// K 0..15, lanes 16-31 hold K 16..31.
static __device__ __forceinline__ int a_kofs(int j, int half) {
  return j + half * 16;
}
// Identity within a 32-column block: column k stored at position k so an
// A fragment (K 16*half..16*half+15) is 16 contiguous halves.
static __device__ __forceinline__ int perm32(int off) {
  return off;
}

static __device__ __forceinline__ v16h a_from_f4(const float* bp, int half) {
  float4 u0 = *(const float4*)(bp + half * 16);
  float4 u1 = *(const float4*)(bp + half * 16 + 4);
  float4 u2 = *(const float4*)(bp + half * 16 + 8);
  float4 u3 = *(const float4*)(bp + half * 16 + 12);
  _Float16 av[16] = {
    (_Float16)u0.x, (_Float16)u0.y, (_Float16)u0.z, (_Float16)u0.w,
    (_Float16)u1.x, (_Float16)u1.y, (_Float16)u1.z, (_Float16)u1.w,
    (_Float16)u2.x, (_Float16)u2.y, (_Float16)u2.z, (_Float16)u2.w,
    (_Float16)u3.x, (_Float16)u3.y, (_Float16)u3.z, (_Float16)u3.w };
  v16h a; __builtin_memcpy(&a, av, sizeof(a));
  return a;
}

static __device__ __forceinline__ v16h a_zero() {
  v16h a = (v16h)(_Float16)0.f;
  return a;
}

// --------------------------- weight pre-swizzle ----------------------------
// frags[kt][nt][lane][16 halves]; B 32x16 frag, this code's contiguous K (same as A): lane<16 -> K 0..15,
// lane>=16 -> K 16..31, col = nt*16 + lane%16.
__global__ __launch_bounds__(32) void smg_prep_w(const float* __restrict__ w,
                                                 _Float16* __restrict__ f,
                                                 int K, int N, int NT) {
  int kt = blockIdx.x / NT, nt = blockIdx.x % NT;
  int lane = threadIdx.x, half = lane >> 4, l16 = lane & 15;
  int n = nt * 16 + l16;
  _Float16* dst = f + ((size_t)(kt * NT + nt) * 32 + lane) * 16;
#pragma unroll
  for (int j = 0; j < 16; ++j) {
    int k = kt * 32 + half * 16 + j;
    float v = (k < K && n < N) ? w[(size_t)k * N + n] : 0.f;
    dst[j] = (_Float16)v;
  }
}

__global__ void smg_norm2(const float* __restrict__ f, const float* __restrict__ mu,
                          const float* __restrict__ sd, float* __restrict__ o, int n) {
  int i = blockIdx.x * blockDim.x + threadIdx.x;
  if (i < n * 2) { int c = i & 1; o[i] = (f[i] - mu[c]) / sd[c]; }
}

__global__ void smg_zero_hi(float* __restrict__ buf, int rows) {
  int i = blockIdx.x * blockDim.x + threadIdx.x;
  if (i < rows * 128) buf[(size_t)(i >> 7) * 256 + 128 + (i & 127)] = 0.f;
}

// ------------------------- fused MLP + LayerNorm (M=32) --------------------
#define HSTR 144   // padded hidden row stride (halves): 288B rows, 32B aligned
#define OSTR 132   // padded output row stride (floats)

__global__ __launch_bounds__(32) void smg_mlp_ln(
    const float* __restrict__ p0, int p0s, int c0, const int* __restrict__ g0,
    const float* __restrict__ p1, int p1s, int c1, const int* __restrict__ g1,
    const float* __restrict__ p2, int p2s, int c2,
    const float* __restrict__ sub, const float* __restrict__ dvd,
    int DIN, int KT1,
    const _Float16* __restrict__ w1f, const float* __restrict__ b1,
    const _Float16* __restrict__ w2f, const float* __restrict__ b2,
    const float* __restrict__ gam, const float* __restrict__ bet,
    const float* __restrict__ res, int ress,
    float* __restrict__ out, int outs,
    float* __restrict__ agg, const int* __restrict__ scat) {
  __shared__ _Float16 hst[32 * HSTR];
  __shared__ float ost[32 * OSTR];
  __shared__ float mu_s[32], rs_s[32];
  const int lane = threadIdx.x, half = lane >> 4, l16 = lane & 15;
  const int row0 = blockIdx.x * 32;
  const int growA = row0 + l16;            // M-subtile 0 row owned by this lane
  const int growB = row0 + 16 + l16;       // M-subtile 1 row
  const int r0a = g0 ? g0[growA] : growA, r0b = g0 ? g0[growB] : growB;
  const int r1a = g1 ? g1[growA] : growA, r1b = g1 ? g1[growB] : growB;
  const float* rp0a = p0 ? p0 + (size_t)r0a * p0s : nullptr;
  const float* rp0b = p0 ? p0 + (size_t)r0b * p0s : nullptr;
  const float* rp1a = p1 ? p1 + (size_t)r1a * p1s : nullptr;
  const float* rp1b = p1 ? p1 + (size_t)r1b * p1s : nullptr;
  const float* rp2a = p2 ? p2 + (size_t)growA * p2s : nullptr;
  const float* rp2b = p2 ? p2 + (size_t)growB * p2s : nullptr;
  const int csum = c0 + c1 + c2;

  const v8f vz = {};
  v8f acc0[8], acc1[8];
#pragma unroll
  for (int nt = 0; nt < 8; ++nt) { acc0[nt] = vz; acc1[nt] = vz; }

  // ---- GEMM1: (32 x DIN) x (DIN x 128) ----
  for (int kt = 0; kt < KT1; ++kt) {
    const int kbase = kt * 32;
    // uniform part resolution; fast path if whole 32-K window is in one part
    const float *bpA = nullptr, *bpB = nullptr;
    if (!sub) {
      if (kbase + 32 <= c0)                            { bpA = rp0a + kbase;            bpB = rp0b + kbase; }
      else if (kbase >= c0 && kbase + 32 <= c0 + c1)   { bpA = rp1a + (kbase - c0);     bpB = rp1b + (kbase - c0); }
      else if (kbase >= c0 + c1 && kbase + 32 <= csum) { bpA = rp2a + (kbase - c0 - c1); bpB = rp2b + (kbase - c0 - c1); }
    }
    v16h a0, a1;
    if (bpA) {
      a0 = a_from_f4(bpA, half);
      a1 = a_from_f4(bpB, half);
    } else {
      _Float16 av0[16], av1[16];
#pragma unroll
      for (int j = 0; j < 16; ++j) {
        int k = kbase + a_kofs(j, half);
        float v0 = 0.f, v1 = 0.f;
        if (k < DIN) {
          if (k < c0)            { v0 = rp0a[k];           v1 = rp0b[k]; }
          else if (k < c0 + c1)  { v0 = rp1a[k - c0];      v1 = rp1b[k - c0]; }
          else                   { v0 = rp2a[k - c0 - c1]; v1 = rp2b[k - c0 - c1]; }
          if (sub) { float s = sub[k], d = dvd[k]; v0 = (v0 - s) / d; v1 = (v1 - s) / d; }
        }
        av0[j] = (_Float16)v0; av1[j] = (_Float16)v1;
      }
      __builtin_memcpy(&a0, av0, sizeof(a0));
      __builtin_memcpy(&a1, av1, sizeof(a1));
    }
    const _Float16* wb = w1f + ((size_t)kt * 8) * 32 * 16 + (size_t)lane * 16;
#pragma unroll
    for (int g4 = 0; g4 < 2; ++g4) {
      v16h bfr[4];
#pragma unroll
      for (int q = 0; q < 4; ++q)
        __builtin_memcpy(&bfr[q], wb + (size_t)(g4 * 4 + q) * 32 * 16, sizeof(v16h));
#pragma unroll
      for (int q = 0; q < 4; ++q) {
        acc0[g4 * 4 + q] = WMMA(a0, bfr[q], acc0[g4 * 4 + q]);
        acc1[g4 * 4 + q] = WMMA(a1, bfr[q], acc1[g4 * 4 + q]);
      }
    }
  }
  // bias + relu -> LDS f16, unpermuted (C layout: VGPR v -> row v+half*8)
#pragma unroll
  for (int nt = 0; nt < 8; ++nt)
#pragma unroll
    for (int v = 0; v < 8; ++v) {
      int r = v + half * 8, c = nt * 16 + l16;
      int pos = (c & ~31) + perm32(c & 31);
      float h0 = acc0[nt][v] + b1[c];
      float h1 = acc1[nt][v] + b1[c];
      hst[r * HSTR + pos]        = (_Float16)(h0 > 0.f ? h0 : 0.f);
      hst[(16 + r) * HSTR + pos] = (_Float16)(h1 > 0.f ? h1 : 0.f);
    }
  __syncthreads();

  // ---- GEMM2: (32 x 128) x (128 x 128); A frag = one 32B LDS load ----
#pragma unroll
  for (int nt = 0; nt < 8; ++nt) { acc0[nt] = vz; acc1[nt] = vz; }
#pragma unroll
  for (int kt = 0; kt < 4; ++kt) {
    v16h a0 = *(const v16h*)(hst + l16 * HSTR + kt * 32 + half * 16);
    v16h a1 = *(const v16h*)(hst + (16 + l16) * HSTR + kt * 32 + half * 16);
    const _Float16* wb = w2f + ((size_t)kt * 8) * 32 * 16 + (size_t)lane * 16;
#pragma unroll
    for (int g4 = 0; g4 < 2; ++g4) {
      v16h bfr[4];
#pragma unroll
      for (int q = 0; q < 4; ++q)
        __builtin_memcpy(&bfr[q], wb + (size_t)(g4 * 4 + q) * 32 * 16, sizeof(v16h));
#pragma unroll
      for (int q = 0; q < 4; ++q) {
        acc0[g4 * 4 + q] = WMMA(a0, bfr[q], acc0[g4 * 4 + q]);
        acc1[g4 * 4 + q] = WMMA(a1, bfr[q], acc1[g4 * 4 + q]);
      }
    }
  }
#pragma unroll
  for (int nt = 0; nt < 8; ++nt)
#pragma unroll
    for (int v = 0; v < 8; ++v) {
      int r = v + half * 8, c = nt * 16 + l16;
      ost[r * OSTR + c]        = acc0[nt][v] + b2[c];
      ost[(16 + r) * OSTR + c] = acc1[nt][v] + b2[c];
    }
  __syncthreads();
  {
    float s = 0.f, s2 = 0.f;
    for (int c = 0; c < 128; ++c) { float h = ost[lane * OSTR + c]; s += h; s2 += h * h; }
    float m = s * (1.f / 128.f);
    mu_s[lane] = m;
    rs_s[lane] = rsqrtf(s2 * (1.f / 128.f) - m * m + 1e-5f);
  }
  __syncthreads();
  for (int i = lane; i < 32 * 128; i += 32) {
    int r = i >> 7, c = i & 127, gr = row0 + r;
    float v = (ost[r * OSTR + c] - mu_s[r]) * rs_s[r] * gam[c] + bet[c];
    if (res) v += res[(size_t)gr * ress + c];
    out[(size_t)gr * outs + c] = v;
    if (agg) atomicAdd(&agg[(size_t)scat[gr] * 128 + c], v);
  }
}

// ------------------------------- decoder (M=32) ----------------------------
__global__ __launch_bounds__(32) void smg_dec(
    const float* __restrict__ xin, int xs, int DIN, int KT1,
    const _Float16* __restrict__ w1f, const float* __restrict__ b1,
    const float* __restrict__ w2, const float* __restrict__ b2,
    float* __restrict__ out) {
  __shared__ float ost[32 * OSTR];
  const int lane = threadIdx.x, half = lane >> 4, l16 = lane & 15;
  const int row0 = blockIdx.x * 32;
  const float* rpA = xin + (size_t)(row0 + l16) * xs;
  const float* rpB = xin + (size_t)(row0 + 16 + l16) * xs;
  const v8f vz = {};
  v8f acc0[8], acc1[8];
#pragma unroll
  for (int nt = 0; nt < 8; ++nt) { acc0[nt] = vz; acc1[nt] = vz; }
  for (int kt = 0; kt < KT1; ++kt) {
    v16h a0 = a_from_f4(rpA + kt * 32, half);  // DIN multiple of 32, one part
    v16h a1 = a_from_f4(rpB + kt * 32, half);
    const _Float16* wb = w1f + ((size_t)kt * 8) * 32 * 16 + (size_t)lane * 16;
#pragma unroll
    for (int g4 = 0; g4 < 2; ++g4) {
      v16h bfr[4];
#pragma unroll
      for (int q = 0; q < 4; ++q)
        __builtin_memcpy(&bfr[q], wb + (size_t)(g4 * 4 + q) * 32 * 16, sizeof(v16h));
#pragma unroll
      for (int q = 0; q < 4; ++q) {
        acc0[g4 * 4 + q] = WMMA(a0, bfr[q], acc0[g4 * 4 + q]);
        acc1[g4 * 4 + q] = WMMA(a1, bfr[q], acc1[g4 * 4 + q]);
      }
    }
  }
#pragma unroll
  for (int nt = 0; nt < 8; ++nt)
#pragma unroll
    for (int v = 0; v < 8; ++v) {
      int r = v + half * 8, c = nt * 16 + l16;
      float h0 = acc0[nt][v] + b1[c];
      float h1 = acc1[nt][v] + b1[c];
      ost[r * OSTR + c]        = h0 > 0.f ? h0 : 0.f;
      ost[(16 + r) * OSTR + c] = h1 > 0.f ? h1 : 0.f;
    }
  __syncthreads();
  {
    float a0 = b2[0], a1 = b2[1];
    for (int k = 0; k < 128; ++k) {
      float h = ost[lane * OSTR + k];
      a0 += h * w2[k * 2 + 0];
      a1 += h * w2[k * 2 + 1];
    }
    out[(size_t)(row0 + lane) * 2 + 0] = a0;
    out[(size_t)(row0 + lane) * 2 + 1] = a1;
  }
}

// --------------------- eigenbroadcast: V(n,64) x spx(64,128) ---------------
#define BBS 80   // Bs column stride (halves): 160B, 32B aligned
__global__ __launch_bounds__(32) void smg_broadcast(
    const float* __restrict__ V, const float* __restrict__ spx_in,
    float* __restrict__ x_out, int npg) {
  const int tpg = (npg + 15) / 16;
  const int g = blockIdx.x / tpg, t = blockIdx.x % tpg;
  const int lane = threadIdx.x, half = lane >> 4, l16 = lane & 15;
  __shared__ _Float16 Bs[128 * BBS];   // column-major: Bs[col*BBS + k]
  for (int i = lane; i < 64 * 128; i += 32) {
    int k = i >> 7, c = i & 127;
    Bs[c * BBS + k] = (_Float16)spx_in[(size_t)(g * 64 + k) * 256 + c];
  }
  __syncthreads();
  const int ng = t * 16 + l16;            // node index inside group (A row)
  const v8f vz = {};
  v8f acc[8];
#pragma unroll
  for (int nt = 0; nt < 8; ++nt) acc[nt] = vz;
#pragma unroll
  for (int kt = 0; kt < 2; ++kt) {
    v16h a = (ng < npg) ? a_from_f4(V + (size_t)(g * npg + ng) * 64 + kt * 32, half)
                        : a_zero();
#pragma unroll
    for (int nt = 0; nt < 8; ++nt) {
      v16h b = *(const v16h*)(Bs + (nt * 16 + l16) * BBS + kt * 32 + half * 16);
      acc[nt] = WMMA(a, b, acc[nt]);
    }
  }
#pragma unroll
  for (int nt = 0; nt < 8; ++nt)
#pragma unroll
    for (int v = 0; v < 8; ++v) {
      int rr = t * 16 + v + half * 8;
      if (rr < npg)
        x_out[(size_t)(g * npg + rr) * 256 + 128 + nt * 16 + l16] = acc[nt][v];
    }
}

// ------------------- eigenpool: V^T(64,n) x x(n,128), K-split --------------
#define PBS 48   // column stride (halves): 96B, 32B aligned
__global__ __launch_bounds__(32) void smg_pool(
    const float* __restrict__ V, const float* __restrict__ x_in,
    float* __restrict__ spx_out, int npg, int chunks) {
  const int per_g = 4 * chunks;
  const int g = blockIdx.x / per_g, rem = blockIdx.x % per_g;
  const int mt = rem / chunks, ch = rem % chunks;
  const int m0 = mt * 16;
  const int kchunk = (npg + chunks - 1) / chunks;
  const int kbeg = ch * kchunk;
  const int kend = (kbeg + kchunk < npg) ? kbeg + kchunk : npg;
  const int lane = threadIdx.x, half = lane >> 4, l16 = lane & 15;
  __shared__ _Float16 Bs[128 * PBS];   // x block, column-major: Bs[col*PBS + k]
  __shared__ _Float16 Vs[16 * PBS];    // V^T block, rows m, K contiguous
  const v8f vz = {};
  v8f acc[8];
#pragma unroll
  for (int nt = 0; nt < 8; ++nt) acc[nt] = vz;
  for (int k0 = kbeg; k0 < kend; k0 += 32) {
    for (int i = lane; i < 32 * 128; i += 32) {
      int r = i >> 7, c = i & 127, k = k0 + r;
      Bs[c * PBS + r] =
          (k < kend) ? (_Float16)x_in[(size_t)(g * npg + k) * 256 + c] : (_Float16)0.f;
    }
    for (int i = lane; i < 32 * 16; i += 32) {
      int r = i >> 4, m = i & 15, k = k0 + r;  // r = node offset, m = eig index
      float v = (k < kend) ? V[(size_t)(g * npg + k) * 64 + m0 + m] : 0.f;
      Vs[m * PBS + perm32(r)] = (_Float16)v;
    }
    __syncthreads();
    v16h a = *(const v16h*)(Vs + l16 * PBS + half * 16);
#pragma unroll
    for (int nt = 0; nt < 8; ++nt) {
      v16h b = *(const v16h*)(Bs + (nt * 16 + l16) * PBS + half * 16);
      acc[nt] = WMMA(a, b, acc[nt]);
    }
    __syncthreads();
  }
#pragma unroll
  for (int nt = 0; nt < 8; ++nt)
#pragma unroll
    for (int v = 0; v < 8; ++v)
      atomicAdd(&spx_out[(size_t)(g * 64 + m0 + v + half * 8) * 256 + 128 + nt * 16 + l16],
                acc[nt][v]);
}

// ===========================================================================
extern "C" void kernel_launch(void* const* d_in, const int* in_sizes, int n_in,
                              void* d_out, int out_size, void* d_ws, size_t ws_size,
                              hipStream_t stream) {
  const int NN = 40000, NE = 120000, G = 8, SPN = 512, SPE = 32768;
  const int NPG = NN / G;

  const float* x    = (const float*)d_in[0];
  const int*   ei   = (const int*)d_in[1];     // [2, NE]: row0=src, row1=dst
  const float* ea   = (const float*)d_in[2];
  const float* forc = (const float*)d_in[3];
  const float* spx0 = (const float*)d_in[4];
  const int*   spei = (const int*)d_in[5];
  const float* spea = (const float*)d_in[6];
  const float* V    = (const float*)d_in[7];
  const float* mx = (const float*)d_in[9],  *sx = (const float*)d_in[10];
  const float* me = (const float*)d_in[11], *se = (const float*)d_in[12];
  const float* mf = (const float*)d_in[13], *sf = (const float*)d_in[14];

  // params flattened jax-tree style after the 15 arrays:
  // dec(b1,b2,w1,w2), edge_enc(b1,b2,beta,g,w1,w2), node_enc(6),
  // proc[3]{edge(6),node(6)}, sp_proc[3]{...}, spedge_enc(6), spnode_enc(6)
  struct MlpP { const float *w1, *b1, *w2, *b2, *g, *be; };
  auto LN6 = [&](int base) -> MlpP {
    return MlpP{ (const float*)d_in[base + 4], (const float*)d_in[base + 0],
                 (const float*)d_in[base + 5], (const float*)d_in[base + 1],
                 (const float*)d_in[base + 3], (const float*)d_in[base + 2] };
  };
  MlpP dec{ (const float*)d_in[17], (const float*)d_in[15],
            (const float*)d_in[18], (const float*)d_in[16], nullptr, nullptr };
  MlpP edge_enc = LN6(19), node_enc = LN6(25);
  MlpP proc_e[3], proc_n[3], sp_e[3], sp_n[3];
  for (int i = 0; i < 3; ++i) { proc_e[i] = LN6(31 + 12 * i); proc_n[i] = LN6(37 + 12 * i); }
  for (int i = 0; i < 3; ++i) { sp_e[i]  = LN6(67 + 12 * i); sp_n[i]  = LN6(73 + 12 * i); }
  MlpP spedge_enc = LN6(103), spnode_enc = LN6(109);

  // ---- workspace carve-out (256B aligned) ----
  char* wsb = (char*)d_ws; size_t off = 0;
  auto allocF = [&](size_t n) -> float* {
    float* p = (float*)(wsb + off); off = (off + n * 4 + 255) & ~(size_t)255; return p; };
  auto allocH = [&](size_t n) -> _Float16* {
    _Float16* p = (_Float16*)(wsb + off); off = (off + n * 2 + 255) & ~(size_t)255; return p; };

  float* xA   = allocF((size_t)NN * 256);
  float* xB   = allocF((size_t)NN * 256);
  float* spxA = allocF((size_t)SPN * 256);
  float* spxB = allocF((size_t)SPN * 256);
  float* e    = allocF((size_t)NE * 128);
  float* spe  = allocF((size_t)SPE * 128);
  float* agg  = allocF((size_t)NN * 128);
  float* spag = allocF((size_t)SPN * 128);
  float* fn   = allocF((size_t)NN * 2);

  struct WF { _Float16 *w1, *w2; int KT1; };
  auto mkWF = [&](int din) -> WF {
    WF w; w.KT1 = (din + 31) / 32;
    w.w1 = allocH((size_t)w.KT1 * 8 * 32 * 16);
    w.w2 = allocH((size_t)4 * 8 * 32 * 16);
    return w;
  };
  auto prep = [&](const float* w, int Kd, _Float16* f, int KT) {
    smg_prep_w<<<dim3(KT * 8), dim3(32), 0, stream>>>(w, f, Kd, 128, 8);
  };
  auto prepBoth = [&](const MlpP& P, int din) -> WF {
    WF w = mkWF(din);
    prep(P.w1, din, w.w1, w.KT1);
    prep(P.w2, 128, w.w2, 4);
    return w;
  };

  WF wf_ne = prepBoth(node_enc, 11);
  WF wf_ee = prepBoth(edge_enc, 3);
  WF wf_sn = prepBoth(spnode_enc, 11);
  WF wf_se = prepBoth(spedge_enc, 3);
  WF wf_pe[3], wf_pn[3], wf_qe[3], wf_qn[3];
  for (int i = 0; i < 3; ++i) {
    wf_pe[i] = prepBoth(proc_e[i], 640);   // 5H
    wf_pn[i] = prepBoth(proc_n[i], 258);   // 2H + DF
    wf_qe[i] = prepBoth(sp_e[i], 640);
    wf_qn[i] = prepBoth(sp_n[i], 256);     // 2H
  }
  WF wf_dc = mkWF(256);
  prep(dec.w1, 256, wf_dc.w1, wf_dc.KT1); // dec.w2 (128x2) used in f32 directly

  auto mlp = [&](int nrows,
                 const float* p0, int p0s, int c0, const int* g0,
                 const float* p1, int p1s, int c1, const int* g1,
                 const float* p2, int p2s, int c2,
                 const float* sub, const float* dvd,
                 int DIN, const WF& wf, const MlpP& P,
                 const float* res, int ress, float* out, int outs,
                 float* aggp, const int* scat) {
    smg_mlp_ln<<<dim3(nrows / 32), dim3(32), 0, stream>>>(
        p0, p0s, c0, g0, p1, p1s, c1, g1, p2, p2s, c2, sub, dvd,
        DIN, wf.KT1, wf.w1, P.b1, wf.w2, P.b2, P.g, P.be,
        res, ress, out, outs, aggp, scat);
  };

  // ---- f normalization ----
  smg_norm2<<<dim3((NN * 2 + 255) / 256), dim3(256), 0, stream>>>(forc, mf, sf, fn, NN);

  // ---- encoders ----
  mlp(NN, x, 11, 11, nullptr, nullptr, 0, 0, nullptr, nullptr, 0, 0,
      mx, sx, 11, wf_ne, node_enc, nullptr, 0, xA, 256, nullptr, nullptr);
  mlp(NE, ea, 3, 3, nullptr, nullptr, 0, 0, nullptr, nullptr, 0, 0,
      me, se, 3, wf_ee, edge_enc, nullptr, 0, e, 128, nullptr, nullptr);
  mlp(SPN, spx0, 11, 11, nullptr, nullptr, 0, 0, nullptr, nullptr, 0, 0,
      nullptr, nullptr, 11, wf_sn, spnode_enc, nullptr, 0, spxA, 256, nullptr, nullptr);
  mlp(SPE, spea, 3, 3, nullptr, nullptr, 0, 0, nullptr, nullptr, 0, 0,
      nullptr, nullptr, 3, wf_se, spedge_enc, nullptr, 0, spe, 128, nullptr, nullptr);

  const int TPG = (NPG + 15) / 16, CH = 8;
  // ---- initial spectral concat: xA[:,128:]=V@spxA[:,:128]; spxA[:,128:]=V^T@xA[:,:128]
  smg_zero_hi<<<dim3((SPN * 128 + 255) / 256), dim3(256), 0, stream>>>(spxA, SPN);
  smg_pool<<<dim3(G * 4 * CH), dim3(32), 0, stream>>>(V, xA, spxA, NPG, CH);
  smg_broadcast<<<dim3(G * TPG), dim3(32), 0, stream>>>(V, spxA, xA, NPG);

  // ---- processor layers ----
  for (int i = 0; i < 3; ++i) {
    float* in_x  = (i % 2 == 0) ? xA : xB;
    float* out_x = (i % 2 == 0) ? xB : xA;
    float* in_s  = (i % 2 == 0) ? spxA : spxB;
    float* out_s = (i % 2 == 0) ? spxB : spxA;
    hipMemsetAsync(agg, 0, (size_t)NN * 128 * 4, stream);
    hipMemsetAsync(spag, 0, (size_t)SPN * 128 * 4, stream);
    // edge MLP: msg=[x[dst], x[src], e] -> e (in place, +residual), scatter agg
    mlp(NE, in_x, 256, 256, ei + NE, in_x, 256, 256, ei, e, 128, 128,
        nullptr, nullptr, 640, wf_pe[i], proc_e[i], e, 128, e, 128, agg, ei + NE);
    // node MLP: [h=x[:, :128], agg, f] -> out_x[:, :128] (+h residual)
    mlp(NN, in_x, 256, 128, nullptr, agg, 128, 128, nullptr, fn, 2, 2,
        nullptr, nullptr, 258, wf_pn[i], proc_n[i], in_x, 256, out_x, 256,
        nullptr, nullptr);
    // sparse edge MLP
    mlp(SPE, in_s, 256, 256, spei + SPE, in_s, 256, 256, spei, spe, 128, 128,
        nullptr, nullptr, 640, wf_qe[i], sp_e[i], spe, 128, spe, 128, spag, spei + SPE);
    // sparse node MLP
    mlp(SPN, in_s, 256, 128, nullptr, spag, 128, 128, nullptr, nullptr, 0, 0,
        nullptr, nullptr, 256, wf_qn[i], sp_n[i], in_s, 256, out_s, 256,
        nullptr, nullptr);
    // spectral concat with OLD x / OLD spx truncated to 128 cols
    smg_zero_hi<<<dim3((SPN * 128 + 255) / 256), dim3(256), 0, stream>>>(out_s, SPN);
    smg_pool<<<dim3(G * 4 * CH), dim3(32), 0, stream>>>(V, in_x, out_s, NPG, CH);
    smg_broadcast<<<dim3(G * TPG), dim3(32), 0, stream>>>(V, in_s, out_x, NPG);
  }

  // ---- decoder: x(final=xB) -> d_out (NN x 2) ----
  smg_dec<<<dim3(NN / 32), dim3(32), 0, stream>>>(
      xB, 256, 256, wf_dc.KT1, wf_dc.w1, dec.b1, dec.w2, dec.b2, (float*)d_out);
}
